// multiGATModelAE_50697793962665
// MI455X (gfx1250) — matrix operand order repaired and checked by reading, not compiled, not run on hardware
//
#include <hip/hip_runtime.h>

typedef __attribute__((ext_vector_type(16))) _Float16 v16h;
typedef __attribute__((ext_vector_type(8)))  _Float16 v8h;
typedef __attribute__((ext_vector_type(8)))  float    v8f;

#define NN      4096
#define NFEAT   1024
#define NHID    64
#define NHEADS  8
#define NCLASS  8
#define NBLOCKS 3
#define ALPHA   0.1f

// ---------------------------------------------------------------- helpers

__device__ __forceinline__ v8f zero8() {
  v8f z;
#pragma unroll
  for (int i = 0; i < 8; ++i) z[i] = 0.f;
  return z;
}

__device__ __forceinline__ float lrelu(float v) { return v > 0.f ? v : ALPHA * v; }

// A fragment (16x32 f16, row-major source, ld in halves).
// Lane l (l&15 = row M); K chunks {kb..kb+7} and {kb+16..kb+23}, kb=(l>>4)*8.
__device__ __forceinline__ v16h load_a_frag(const _Float16* A, int lda, int row0,
                                            int k0, int lane) {
  int r = lane & 15, kb = (lane >> 4) * 8;
  const _Float16* p = A + (long)(row0 + r) * lda + k0 + kb;
  v8h lo = *(const v8h*)p;
  v8h hi = *(const v8h*)(p + 16);
  v16h a;
#pragma unroll
  for (int i = 0; i < 8; ++i) { a[i] = lo[i]; a[i + 8] = hi[i]; }
  return a;
}

// B fragment (32x16 f16) from BT stored row-major [N,K] (ld in halves).
// Lane l: column N = l&15; K chunks {kb..kb+7} and {kb+16..kb+23}, kb=(l>>4)*8.
__device__ __forceinline__ v16h load_b_frag(const _Float16* BT, int ldbt, int n0,
                                            int k0, int lane) {
  int c = lane & 15, ks = (lane >> 4) * 8;
  const _Float16* p = BT + (long)(n0 + c) * ldbt + k0 + ks;
  v8h lo = *(const v8h*)p;
  v8h hi = *(const v8h*)(p + 16);
  v16h b;
#pragma unroll
  for (int i = 0; i < 8; ++i) { b[i] = lo[i]; b[i + 8] = hi[i]; }
  return b;
}

__device__ __forceinline__ v8f wmma32(v16h a, v16h b, v8f c) {
  return __builtin_amdgcn_wmma_f32_16x16x32_f16(false, a, false, b, (short)0, c,
                                                false, false);
}

// ---------------------------------------------------------------- utility kernels

__global__ void conv_f16_k(const float* __restrict__ in, _Float16* __restrict__ out,
                           long n) {
  long i = (long)blockIdx.x * blockDim.x + threadIdx.x;
  if (i < n) out[i] = (_Float16)in[i];
}

// TT[c*rows + r] = (f16)(T[r*cols + c] * (dinv ? dinv[r] : 1))   (batched via grid.y)
__global__ void scale_transpose_k(const float* __restrict__ T,
                                  const float* __restrict__ dinv,
                                  _Float16* __restrict__ TT, int rows, int cols,
                                  long sT, long sTT) {
  long i = (long)blockIdx.x * blockDim.x + threadIdx.x;
  long total = (long)rows * cols;
  if (i >= total) return;
  int b = blockIdx.y;
  int r = (int)(i / cols), c = (int)(i % cols);
  float v = T[b * sT + i];
  if (dinv) v *= dinv[r];
  TT[b * sTT + (long)c * rows + r] = (_Float16)v;
}

// W_cls [64,8] -> padded BT [16,64], rows 8..15 zero
__global__ void wcls_pad_k(const float* __restrict__ Wcls,
                           _Float16* __restrict__ WclsT) {
  int i = blockIdx.x * blockDim.x + threadIdx.x;
  if (i >= 16 * NHID) return;
  int n = i / NHID, k = i % NHID;
  WclsT[i] = (n < NCLASS) ? (_Float16)Wcls[(long)k * NCLASS + n] : (_Float16)0.f;
}

__global__ void degree_rsqrt_k(const int* __restrict__ adj, float* __restrict__ dinv) {
  __shared__ int red[256];
  int i = blockIdx.x, t = threadIdx.x;
  int s = 0;
  const int* arow = adj + (long)i * NN;
  for (int j = t; j < NN; j += 256) s += arow[j];
  red[t] = s;
  __syncthreads();
  for (int o = 128; o > 0; o >>= 1) {
    if (t < o) red[t] += red[t + o];
    __syncthreads();
  }
  if (t == 0) dinv[i] = rsqrtf((float)red[0]);
}

// f1 = Wh @ a[:64], f2 = Wh @ a[64:]   (batched via grid.y)
__global__ void f12_k(const float* __restrict__ WhB, long sWh,
                      const float* __restrict__ aB, long sA,
                      float* __restrict__ f12B, long sF) {
  int row = blockIdx.x * blockDim.x + threadIdx.x;
  if (row >= NN) return;
  int b = blockIdx.y;
  const float* wh = WhB + b * sWh + (long)row * NHID;
  const float* av = aB + b * sA;
  float f1 = 0.f, f2 = 0.f;
#pragma unroll
  for (int k = 0; k < NHID; ++k) {
    float w = wh[k];
    f1 += w * av[k];
    f2 += w * av[NHID + k];
  }
  f12B[b * sF + row] = f1;
  f12B[b * sF + NN + row] = f2;
}

// ---------------------------------------------------------------- WMMA GEMM
// C[M,N] = A[M,K] (f16, row-major) x BT[N,K]^T (f16). One 16x16 tile per wave.
__global__ void gemm_f16(const _Float16* __restrict__ A,
                         const _Float16* __restrict__ BT, float* __restrict__ C,
                         int N, int K, int lda, int ldbt, int ldc, long sA, long sB,
                         long sC) {
  int lane = threadIdx.x & 31, wave = threadIdx.x >> 5;
  int ntile = blockIdx.y * 4 + wave;
  if (ntile * 16 >= N) return;
  long b = blockIdx.z;
  A += b * sA;
  BT += b * sB;
  C += b * sC;
  int row0 = blockIdx.x * 16, n0 = ntile * 16;
  v8f acc = zero8();
  for (int k0 = 0; k0 < K; k0 += 32) {
    v16h a = load_a_frag(A, lda, row0, k0, lane);
    v16h bb = load_b_frag(BT, ldbt, n0, k0, lane);
    acc = wmma32(a, bb, acc);
  }
  int half = lane >> 4, c = lane & 15;
#pragma unroll
  for (int r = 0; r < 8; ++r)
    C[(long)(row0 + r + 8 * half) * ldc + n0 + c] = acc[r];
}

// ---------------------------------------------------------------- attention stats
// m[i] = lrelu(f1[i] + max_{j in mask} f2[j]) (lrelu monotone => exact row max)
// s[i] = sum_{j in mask} exp(lrelu(f1[i]+f2[j]) - m[i])
__global__ void att_stats_k(const int* __restrict__ adj,
                            const float* __restrict__ f12B, long sF,
                            float* __restrict__ msB, long sM) {
  __shared__ float red[256];
  int i = blockIdx.x, t = threadIdx.x, b = blockIdx.y;
  const float* f2 = f12B + b * sF + NN;
  float f1 = f12B[b * sF + i];
  const int* arow = adj + (long)i * NN;
  float mx = -3.0e38f;
  for (int j = t; j < NN; j += 256)
    if (arow[j]) mx = fmaxf(mx, f2[j]);
  red[t] = mx;
  __syncthreads();
  for (int o = 128; o > 0; o >>= 1) {
    if (t < o) red[t] = fmaxf(red[t], red[t + o]);
    __syncthreads();
  }
  float m = lrelu(f1 + red[0]);
  __syncthreads();
  float s = 0.f;
  for (int j = t; j < NN; j += 256)
    if (arow[j]) s += __expf(lrelu(f1 + f2[j]) - m);
  red[t] = s;
  __syncthreads();
  for (int o = 128; o > 0; o >>= 1) {
    if (t < o) red[t] += red[t + o];
    __syncthreads();
  }
  if (t == 0) {
    msB[b * sM + i] = m;
    msB[b * sM + NN + i] = red[0];
  }
}

// ---------------------------------------------------------------- fused softmax x Wh
// One wave per 16-row block; builds P tile in registers as the WMMA A fragment.
__global__ void att_apply_k(const int* __restrict__ adj,
                            const float* __restrict__ f12B, long sF,
                            const float* __restrict__ msB, long sM,
                            const _Float16* __restrict__ WhTB, long sW,
                            _Float16* __restrict__ Hout, int ldh, int do_elu) {
  int lane = threadIdx.x & 31;
  int b = blockIdx.y;
  const float* f2 = f12B + b * sF + NN;
  const float* srow = msB + b * sM + NN;
  const _Float16* WhT = WhTB + b * sW;
  int row0 = blockIdx.x * 16;
  int rA = lane & 15, half = lane >> 4, kb = half * 8;
  float f1v = f12B[b * sF + row0 + rA];
  float mv = msB[b * sM + row0 + rA];
  v8f acc0 = zero8(), acc1 = zero8(), acc2 = zero8(), acc3 = zero8();
  const int* ap0 = adj + (long)(row0 + rA) * NN + kb;
  const float* fp0 = f2 + kb;
  for (int j0 = 0; j0 < NN; j0 += 32) {
    const int* ap = ap0 + j0;
    const float* fp = fp0 + j0;
    v16h pa;
#pragma unroll
    for (int i = 0; i < 8; ++i) {
      float e0 = lrelu(f1v + fp[i]);
      float e1 = lrelu(f1v + fp[16 + i]);
      pa[i] = (_Float16)(ap[i] ? __expf(e0 - mv) : 0.f);
      pa[8 + i] = (_Float16)(ap[16 + i] ? __expf(e1 - mv) : 0.f);
    }
    acc0 = wmma32(pa, load_b_frag(WhT, NN, 0, j0, lane), acc0);
    acc1 = wmma32(pa, load_b_frag(WhT, NN, 16, j0, lane), acc1);
    acc2 = wmma32(pa, load_b_frag(WhT, NN, 32, j0, lane), acc2);
    acc3 = wmma32(pa, load_b_frag(WhT, NN, 48, j0, lane), acc3);
  }
  int c = lane & 15;
  long colbase = (long)b * NHID;
#pragma unroll
  for (int r = 0; r < 8; ++r) {
    int row = row0 + r + 8 * half;
    float inv = 1.f / srow[row];
    float v0 = acc0[r] * inv, v1 = acc1[r] * inv, v2 = acc2[r] * inv,
          v3 = acc3[r] * inv;
    if (do_elu) {
      v0 = v0 > 0.f ? v0 : expm1f(v0);
      v1 = v1 > 0.f ? v1 : expm1f(v1);
      v2 = v2 > 0.f ? v2 : expm1f(v2);
      v3 = v3 > 0.f ? v3 : expm1f(v3);
    }
    _Float16* op = Hout + (long)row * ldh + colbase + c;
    op[0] = (_Float16)v0;
    op[16] = (_Float16)v1;
    op[32] = (_Float16)v2;
    op[48] = (_Float16)v3;
  }
}

// ---------------------------------------------------------------- GCN propagation
// out[i,:] = act(dinv[i] * sum_j adj[i,j] * TT[:,j]), TT pre-scaled by dinv[j].
// mode 0: f16 out [NN,64], relu.  mode 1: f32 out [NN,8], no act.
__global__ void gcn_prop_k(const int* __restrict__ adj,
                           const float* __restrict__ dinv,
                           const _Float16* __restrict__ TT,
                           _Float16* __restrict__ outH, float* __restrict__ outF,
                           int mode) {
  int lane = threadIdx.x & 31;
  int row0 = blockIdx.x * 16;
  int rA = lane & 15, half = lane >> 4, kb = half * 8;
  const int* ap0 = adj + (long)(row0 + rA) * NN + kb;
  v8f acc0 = zero8(), acc1 = zero8(), acc2 = zero8(), acc3 = zero8();
  const _Float16 one = (_Float16)1.f, zro = (_Float16)0.f;
  for (int j0 = 0; j0 < NN; j0 += 32) {
    const int* ap = ap0 + j0;
    v16h pa;
#pragma unroll
    for (int i = 0; i < 8; ++i) {
      pa[i] = ap[i] ? one : zro;
      pa[8 + i] = ap[16 + i] ? one : zro;
    }
    acc0 = wmma32(pa, load_b_frag(TT, NN, 0, j0, lane), acc0);
    if (mode == 0) {
      acc1 = wmma32(pa, load_b_frag(TT, NN, 16, j0, lane), acc1);
      acc2 = wmma32(pa, load_b_frag(TT, NN, 32, j0, lane), acc2);
      acc3 = wmma32(pa, load_b_frag(TT, NN, 48, j0, lane), acc3);
    }
  }
  int c = lane & 15;
#pragma unroll
  for (int r = 0; r < 8; ++r) {
    int row = row0 + r + 8 * half;
    float di = dinv[row];
    if (mode == 0) {
      _Float16* op = outH + (long)row * NHID + c;
      op[0] = (_Float16)fmaxf(acc0[r] * di, 0.f);
      op[16] = (_Float16)fmaxf(acc1[r] * di, 0.f);
      op[32] = (_Float16)fmaxf(acc2[r] * di, 0.f);
      op[48] = (_Float16)fmaxf(acc3[r] * di, 0.f);
    } else {
      if (c < NCLASS) outF[(long)row * NCLASS + c] = acc0[r] * di;
    }
  }
}

// ---------------------------------------------------------------- host driver

extern "C" void kernel_launch(void* const* d_in, const int* in_sizes, int n_in,
                              void* d_out, int out_size, void* d_ws, size_t ws_size,
                              hipStream_t stream) {
  (void)in_sizes; (void)n_in; (void)out_size; (void)ws_size;
  const float* x      = (const float*)d_in[0];
  const int*   adj    = (const int*)d_in[1];
  const float* Wheads = (const float*)d_in[2];
  const float* aheads = (const float*)d_in[3];
  const float* Wout   = (const float*)d_in[4];
  const float* aout   = (const float*)d_in[5];
  const float* Wgcn   = (const float*)d_in[6];
  const float* Wcls   = (const float*)d_in[7];
  float* out = (float*)d_out;

  char* ws = (char*)d_ws;
  size_t off = 0;
  auto alloc = [&](size_t bytes) -> char* {
    char* p = ws + off;
    off += (bytes + 255) & ~(size_t)255;
    return p;
  };
  _Float16* x16   = (_Float16*)alloc((size_t)NN * NFEAT * 2);
  _Float16* WTh   = (_Float16*)alloc((size_t)NHEADS * NHID * NFEAT * 2);
  float*    Wh    = (float*)alloc((size_t)NHEADS * NN * NHID * 4);
  _Float16* WhT   = (_Float16*)alloc((size_t)NHEADS * NHID * NN * 2);
  float*    f12   = (float*)alloc((size_t)9 * 2 * NN * 4);
  float*    ms    = (float*)alloc((size_t)9 * 2 * NN * 4);
  _Float16* hcat  = (_Float16*)alloc((size_t)NN * NHEADS * NHID * 2);
  float*    Wh2   = (float*)alloc((size_t)NN * NHID * 4);
  _Float16* Wh2T  = (_Float16*)alloc((size_t)NHID * NN * 2);
  _Float16* WoutT = (_Float16*)alloc((size_t)NHID * NHEADS * NHID * 2);
  _Float16* h2    = (_Float16*)alloc((size_t)NN * NHID * 2);
  float*    dinv  = (float*)alloc((size_t)NN * 4);
  _Float16* WgcnT = (_Float16*)alloc((size_t)NBLOCKS * NHID * NHID * 2);
  _Float16* WclsT = (_Float16*)alloc((size_t)16 * NHID * 2);
  float*    tbuf  = (float*)alloc((size_t)NN * NHID * 4);
  _Float16* tT    = (_Float16*)alloc((size_t)NHID * NN * 2);
  _Float16* hping = (_Float16*)alloc((size_t)NN * NHID * 2);
  _Float16* hpong = (_Float16*)alloc((size_t)NN * NHID * 2);
  float*    tcls  = (float*)alloc((size_t)NN * 16 * 4);
  _Float16* tclsT = (_Float16*)alloc((size_t)16 * NN * 2);

  const long sL = 2L * NN;  // f12/ms per-layer stride

  // ---- input / weight prep
  {
    long n = (long)NN * NFEAT;
    conv_f16_k<<<dim3((unsigned)((n + 255) / 256)), 256, 0, stream>>>(x, x16, n);
  }
  {
    long n = (long)NFEAT * NHID;
    scale_transpose_k<<<dim3((unsigned)((n + 255) / 256), NHEADS), 256, 0, stream>>>(
        Wheads, nullptr, WTh, NFEAT, NHID, n, n);
  }
  {
    long n = (long)(NHEADS * NHID) * NHID;
    scale_transpose_k<<<dim3((unsigned)((n + 255) / 256), 1), 256, 0, stream>>>(
        Wout, nullptr, WoutT, NHEADS * NHID, NHID, 0, 0);
  }
  {
    long n = (long)NHID * NHID;
    scale_transpose_k<<<dim3((unsigned)((n + 255) / 256), NBLOCKS), 256, 0, stream>>>(
        Wgcn, nullptr, WgcnT, NHID, NHID, n, n);
  }
  wcls_pad_k<<<4, 256, 0, stream>>>(Wcls, WclsT);
  degree_rsqrt_k<<<NN, 256, 0, stream>>>(adj, dinv);

  // ---- per-head Wh = X @ W  (batched WMMA GEMM)
  gemm_f16<<<dim3(NN / 16, 1, NHEADS), 128, 0, stream>>>(
      x16, WTh, Wh, NHID, NFEAT, NFEAT, NFEAT, NHID, 0L, (long)NHID * NFEAT,
      (long)NN * NHID);
  f12_k<<<dim3(NN / 256, NHEADS), 256, 0, stream>>>(Wh, (long)NN * NHID, aheads,
                                                    2L * NHID, f12, sL);
  att_stats_k<<<dim3(NN, NHEADS), 256, 0, stream>>>(adj, f12, sL, ms, sL);
  {
    long n = (long)NN * NHID;
    scale_transpose_k<<<dim3((unsigned)((n + 255) / 256), NHEADS), 256, 0, stream>>>(
        Wh, nullptr, WhT, NN, NHID, n, n);
  }
  att_apply_k<<<dim3(NN / 16, NHEADS), 32, 0, stream>>>(
      adj, f12, sL, ms, sL, WhT, (long)NHID * NN, hcat, NHEADS * NHID, 1);

  // ---- output GAT layer
  gemm_f16<<<dim3(NN / 16, 1, 1), 128, 0, stream>>>(
      hcat, WoutT, Wh2, NHID, NHEADS * NHID, NHEADS * NHID, NHEADS * NHID, NHID,
      0, 0, 0);
  f12_k<<<dim3(NN / 256, 1), 256, 0, stream>>>(Wh2, 0, aout, 0, f12 + 8 * sL, 0);
  att_stats_k<<<dim3(NN, 1), 256, 0, stream>>>(adj, f12 + 8 * sL, 0, ms + 8 * sL, 0);
  {
    long n = (long)NN * NHID;
    scale_transpose_k<<<dim3((unsigned)((n + 255) / 256), 1), 256, 0, stream>>>(
        Wh2, nullptr, Wh2T, NN, NHID, 0, 0);
  }
  att_apply_k<<<dim3(NN / 16, 1), 32, 0, stream>>>(adj, f12 + 8 * sL, 0,
                                                   ms + 8 * sL, 0, Wh2T, 0, h2,
                                                   NHID, 0);

  // ---- GCN blocks: h = relu(a_norm @ (h @ W))
  _Float16* hc = h2;
  _Float16* hn = hping;
  for (int blk = 0; blk < NBLOCKS; ++blk) {
    gemm_f16<<<dim3(NN / 16, 1, 1), 128, 0, stream>>>(
        hc, WgcnT + (long)blk * NHID * NHID, tbuf, NHID, NHID, NHID, NHID, NHID,
        0, 0, 0);
    long n = (long)NN * NHID;
    scale_transpose_k<<<dim3((unsigned)((n + 255) / 256), 1), 256, 0, stream>>>(
        tbuf, dinv, tT, NN, NHID, 0, 0);
    gcn_prop_k<<<dim3(NN / 16, 1), 32, 0, stream>>>(adj, dinv, tT, hn, nullptr, 0);
    hc = hn;
    hn = (hn == hping) ? hpong : hping;
  }

  // ---- classifier head: out = a_norm @ (h @ W_cls)
  gemm_f16<<<dim3(NN / 16, 1, 1), 128, 0, stream>>>(hc, WclsT, tcls, 16, NHID,
                                                    NHID, NHID, 16, 0, 0, 0);
  {
    long n = (long)NN * 16;
    scale_transpose_k<<<dim3((unsigned)((n + 255) / 256), 1), 256, 0, stream>>>(
        tcls, dinv, tclsT, NN, 16, 0, 0);
  }
  gcn_prop_k<<<dim3(NN / 16, 1), 32, 0, stream>>>(adj, dinv, tclsT, nullptr, out, 1);
}
